// EncoderProcessorDecoder_36532991820472
// MI455X (gfx1250) — matrix operand order repaired and checked by reading, not compiled, not run on hardware
//
#include <hip/hip_runtime.h>
#include <cstdint>

typedef unsigned int u32;
typedef __attribute__((ext_vector_type(16))) __bf16 v16bf;
typedef __attribute__((ext_vector_type(8)))  float  v8f;

#define WAIT_DS() asm volatile("s_wait_dscnt 0" ::: "memory")

static constexpr int H = 128;
static constexpr int NNODES = 50000;
static constexpr int NEDGES = 800000;

// LDS sizing (u32 units)
static constexpr int EDGE_WGT_U32 = (384*128 + 128*128 + 128*128) / 2;  // 40960
static constexpr int EDGE_STG_U32 = (16*384)/2 + 64;                    // 3136
static constexpr int NODE_WGT_U32 = (256*128 + 128*128 + 128*128) / 2;  // 32768
static constexpr int NODE_STG_U32 = 2048 + 64;
static constexpr int ENC_WGT_U32  = (32*128 + 128*128 + 128*128) / 2;   // 18432
static constexpr int ENC_STG_U32  = 2048 + 64;
static constexpr int DEC_WGT_U32  = (128*128 + 128*128 + 16*128) / 2;   // 17408
static constexpr int DEC_STG_U32  = 2048 + 64;

static constexpr int EDGE_LDS_BYTES = (EDGE_WGT_U32 + 8*EDGE_STG_U32) * 4; // 264192
static constexpr int NODE_LDS_BYTES = (NODE_WGT_U32 + 8*NODE_STG_U32) * 4; // 198656
static constexpr int ENC_LDS_BYTES  = (ENC_WGT_U32  + 8*ENC_STG_U32)  * 4; // 141312
static constexpr int DEC_LDS_BYTES  = (DEC_WGT_U32  + 8*DEC_STG_U32)  * 4; // 137216

union Frag { v16bf v; uint4 q[2]; };

// One GEMM layer on a 16-row tile: C[16,NT*16] = A[16,K] * W^T + bias.
// A staged in LDS as bf16 [16][K] (pitch aPitch u32). W in LDS as bf16 [N][K].
// A-frag (16-bit, 16x32): lane-half g holds K runs [k0+8g, +7] and [k0+16+8g, +7].
// B-frag: lane-half g holds K runs [k0+8g, +7] and [k0+16+8g, +7] for column n.
template<int NT, bool RELU, bool TO_F32>
__device__ __forceinline__ void gemm_layer(const u32* aStag, int aPitch,
                                           const u32* wT, int K,
                                           const float* bias, int biasN,
                                           u32* outStag, float* outF, int lane)
{
  const int nl = lane & 15;
  const int g  = lane >> 4;
  const int KU = K >> 1;
  v8f acc[NT];
  v8f z = {0.f,0.f,0.f,0.f,0.f,0.f,0.f,0.f};
#pragma unroll
  for (int i = 0; i < NT; ++i) acc[i] = z;

  for (int k0 = 0; k0 < K; k0 += 32) {
    Frag a;
    const u32* ap = aStag + nl * aPitch + (k0 >> 1) + 4 * g;
    a.q[0] = *(const uint4*)(ap);
    a.q[1] = *(const uint4*)(ap + 8);
#pragma unroll
    for (int nt = 0; nt < NT; ++nt) {
      Frag b;
      const u32* wp = wT + (nt*16 + nl) * KU + (k0 >> 1) + 4 * g;
      b.q[0] = *(const uint4*)(wp);
      b.q[1] = *(const uint4*)(wp + 8);
      acc[nt] = __builtin_amdgcn_wmma_f32_16x16x32_bf16(
          false, a.v, false, b.v, (short)0, acc[nt], false, false);
    }
  }
  // C layout: lane-half g: n = nl (+nt*16); VGPR r -> row m = r + 8*g
#pragma unroll
  for (int nt = 0; nt < NT; ++nt) {
    int n = nt*16 + nl;
    float bv = (n < biasN) ? bias[n] : 0.f;
#pragma unroll
    for (int r = 0; r < 8; ++r) {
      int m = r + 8*g;
      float v = acc[nt][r] + bv;
      if constexpr (RELU) v = fmaxf(v, 0.f);
      if constexpr (TO_F32) outF[m * (NT*16) + n] = v;
      else ((__bf16*)outStag)[m * 128 + n] = (__bf16)v;
    }
  }
}

template<int NT2>
__device__ __forceinline__ void mlp3(u32* stag, int p0, int K0,
                                     const u32* w0, const u32* w1, const u32* w2,
                                     const float* b0, const float* b1, const float* b2,
                                     int b2N, int lane)
{
  gemm_layer<8, true, false>(stag, p0, w0, K0, b0, 128, stag, nullptr, lane);
  WAIT_DS();
  gemm_layer<8, true, false>(stag, 64, w1, 128, b1, 128, stag, nullptr, lane);
  WAIT_DS();
  gemm_layer<NT2, false, true>(stag, 64, w2, 128, b2, b2N, nullptr, (float*)stag, lane);
  WAIT_DS();
}

__device__ __forceinline__ void ln_stats(const float* fst, float* muA, float* rsA, int lane)
{
  if (lane < 16) {
    float s = 0.f, s2 = 0.f;
    const float4* rp = (const float4*)(fst + lane * 128);
#pragma unroll
    for (int c = 0; c < 32; ++c) {
      float4 v = rp[c];
      s  += v.x + v.y + v.z + v.w;
      s2 += v.x*v.x + v.y*v.y + v.z*v.z + v.w*v.w;
    }
    float mu  = s * (1.f/128.f);
    float var = s2 * (1.f/128.f) - mu*mu;
    muA[lane] = mu;
    rsA[lane] = rsqrtf(var + 1e-5f);
  }
  WAIT_DS();
}

// ---------------- Encoder: node MLP (16 -> 128 -> 128 -> 128, LN) ----------------
__global__ __launch_bounds__(256) void enc_node_kernel(
    const float* __restrict__ x, float* __restrict__ node_h, __bf16* __restrict__ node_bf,
    const __bf16* __restrict__ wgt, const float* b0, const float* b1, const float* b2,
    const float* lng, const float* lnb, int N)
{
  extern __shared__ u32 lds[];
  const int tid = threadIdx.x, lane = tid & 31, wid = tid >> 5;
  for (int i = tid; i < ENC_WGT_U32/4; i += blockDim.x)
    ((uint4*)lds)[i] = ((const uint4*)wgt)[i];
  __syncthreads();
  const u32* w0 = lds;
  const u32* w1 = lds + (32*128)/2;
  const u32* w2 = w1 + (128*128)/2;
  u32* stag = lds + ENC_WGT_U32 + wid * ENC_STG_U32;
  float* muA = (float*)(stag + 2048);
  float* rsA = muA + 16;

  int64_t n0 = ((int64_t)blockIdx.x * 8 + wid) * 16;
  if (n0 >= N) return;

  __bf16* sb = (__bf16*)stag;  // A0 tile 16x32 (padded K)
  for (int idx = lane; idx < 16*32; idx += 32) {
    int row = idx >> 5, k = idx & 31;
    int64_t nid = n0 + row; if (nid > N-1) nid = N-1;
    float v = (k < 16) ? x[nid*16 + k] : 0.f;
    sb[idx] = (__bf16)v;
  }
  WAIT_DS();
  mlp3<8>(stag, 16, 32, w0, w1, w2, b0, b1, b2, 128, lane);
  float* fst = (float*)stag;
  ln_stats(fst, muA, rsA, lane);
  for (int idx = lane; idx < 2048; idx += 32) {
    int row = idx >> 7, col = idx & 127;
    int64_t nid = n0 + row;
    if (nid < N) {
      float v = (fst[idx] - muA[row]) * rsA[row] * lng[col] + lnb[col];
      node_h[nid*H + col]  = v;
      node_bf[nid*H + col] = (__bf16)v;
    }
  }
}

// ---------------- Encoder: edge MLP (4 -> 128 -> 128 -> 128, LN) ----------------
__global__ __launch_bounds__(256) void enc_edge_kernel(
    const float* __restrict__ eattr, __bf16* __restrict__ edge_bf,
    const __bf16* __restrict__ wgt, const float* b0, const float* b1, const float* b2,
    const float* lng, const float* lnb, int E)
{
  extern __shared__ u32 lds[];
  const int tid = threadIdx.x, lane = tid & 31, wid = tid >> 5;
  for (int i = tid; i < ENC_WGT_U32/4; i += blockDim.x)
    ((uint4*)lds)[i] = ((const uint4*)wgt)[i];
  __syncthreads();
  const u32* w0 = lds;
  const u32* w1 = lds + (32*128)/2;
  const u32* w2 = w1 + (128*128)/2;
  u32* stag = lds + ENC_WGT_U32 + wid * ENC_STG_U32;
  float* muA = (float*)(stag + 2048);
  float* rsA = muA + 16;

  int64_t e0 = ((int64_t)blockIdx.x * 8 + wid) * 16;
  if (e0 >= E) return;

  __bf16* sb = (__bf16*)stag;
  for (int idx = lane; idx < 16*32; idx += 32) {
    int row = idx >> 5, k = idx & 31;
    int64_t eid = e0 + row; if (eid > E-1) eid = E-1;
    float v = (k < 4) ? eattr[eid*4 + k] : 0.f;
    sb[idx] = (__bf16)v;
  }
  WAIT_DS();
  mlp3<8>(stag, 16, 32, w0, w1, w2, b0, b1, b2, 128, lane);
  float* fst = (float*)stag;
  ln_stats(fst, muA, rsA, lane);
  for (int idx = lane; idx < 2048; idx += 32) {
    int row = idx >> 7, col = idx & 127;
    int64_t eid = e0 + row;
    if (eid < E)
      edge_bf[eid*H + col] =
          (__bf16)((fst[idx] - muA[row]) * rsA[row] * lng[col] + lnb[col]);
  }
}

// ------- MP block, edge side: MLP(concat[node_s,node_r,edge]) + LN + residual + scatter-add -------
__global__ __launch_bounds__(256) void edge_block_kernel(
    const __bf16* __restrict__ node_bf, __bf16* __restrict__ edge_bf, float* __restrict__ agg,
    const int* __restrict__ senders, const int* __restrict__ receivers,
    const __bf16* __restrict__ wgt, const float* b0, const float* b1, const float* b2,
    const float* lng, const float* lnb, int E)
{
  extern __shared__ u32 lds[];
  const int tid = threadIdx.x, lane = tid & 31, wid = tid >> 5;
  for (int i = tid; i < EDGE_WGT_U32/4; i += blockDim.x)
    ((uint4*)lds)[i] = ((const uint4*)wgt)[i];
  __syncthreads();
  const u32* w0 = lds;                       // [128][384] bf16
  const u32* w1 = lds + (384*128)/2;         // [128][128]
  const u32* w2 = w1 + (128*128)/2;          // [128][128]
  u32* stag = lds + EDGE_WGT_U32 + wid * EDGE_STG_U32;
  u32* sndA = stag + 3072;
  u32* rcvA = sndA + 16;
  float* muA = (float*)(rcvA + 16);
  float* rsA = muA + 16;

  int64_t e0 = ((int64_t)blockIdx.x * 8 + wid) * 16;
  if (e0 >= E) return;

  if (lane < 16) {
    int64_t e = e0 + lane; if (e > E-1) e = E-1;
    sndA[lane] = (u32)senders[e];
    rcvA[lane] = (u32)receivers[e];
  }
  WAIT_DS();

  // Stage A0 = [node_bf[s] | node_bf[r] | edge_bf[e]]  (16 rows x 384 bf16)
  for (int row = 0; row < 16; ++row) {
    int64_t e = e0 + row; if (e > E-1) e = E-1;
    u32 s = sndA[row], r = rcvA[row];
    uint2 d0 = ((const uint2*)(node_bf + (size_t)s * H))[lane];
    uint2 d1 = ((const uint2*)(node_bf + (size_t)r * H))[lane];
    uint2 d2 = ((const uint2*)(edge_bf + (size_t)e * H))[lane];
    ((uint2*)(stag + row*192      ))[lane] = d0;
    ((uint2*)(stag + row*192 +  64))[lane] = d1;
    ((uint2*)(stag + row*192 + 128))[lane] = d2;
  }
  WAIT_DS();

  mlp3<8>(stag, 192, 384, w0, w1, w2, b0, b1, b2, 128, lane);

  float* fst = (float*)stag;
  ln_stats(fst, muA, rsA, lane);

  for (int idx = lane; idx < 2048; idx += 32) {
    int row = idx >> 7, col = idx & 127;
    int64_t e = e0 + row;
    if (e < E) {
      float en = (fst[idx] - muA[row]) * rsA[row] * lng[col] + lnb[col];
      size_t eo = (size_t)e * H + col;
      edge_bf[eo] = (__bf16)((float)edge_bf[eo] + en);     // residual
      atomicAdd(&agg[(size_t)rcvA[row] * H + col], en);    // segment_sum
    }
  }
}

// ------- MP block, node side: MLP(concat[node, agg]) + LN + residual -------
__global__ __launch_bounds__(256) void node_block_kernel(
    float* __restrict__ node_h, __bf16* __restrict__ node_bf, const float* __restrict__ agg,
    const __bf16* __restrict__ wgt, const float* b0, const float* b1, const float* b2,
    const float* lng, const float* lnb, int N)
{
  extern __shared__ u32 lds[];
  const int tid = threadIdx.x, lane = tid & 31, wid = tid >> 5;
  for (int i = tid; i < NODE_WGT_U32/4; i += blockDim.x)
    ((uint4*)lds)[i] = ((const uint4*)wgt)[i];
  __syncthreads();
  const u32* w0 = lds;                       // [128][256]
  const u32* w1 = lds + (256*128)/2;
  const u32* w2 = w1 + (128*128)/2;
  u32* stag = lds + NODE_WGT_U32 + wid * NODE_STG_U32;
  float* muA = (float*)(stag + 2048);
  float* rsA = muA + 16;

  int64_t n0 = ((int64_t)blockIdx.x * 8 + wid) * 16;
  if (n0 >= N) return;

  for (int row = 0; row < 16; ++row) {
    int64_t nid = n0 + row; if (nid > N-1) nid = N-1;
    uint2 d = ((const uint2*)(node_bf + (size_t)nid * H))[lane];
    ((uint2*)(stag + row*128))[lane] = d;
    float4 f = ((const float4*)(agg + (size_t)nid * H))[lane];
    union { __bf16 h[4]; uint2 u; } pk;
    pk.h[0] = (__bf16)f.x; pk.h[1] = (__bf16)f.y;
    pk.h[2] = (__bf16)f.z; pk.h[3] = (__bf16)f.w;
    ((uint2*)(stag + row*128 + 64))[lane] = pk.u;
  }
  WAIT_DS();

  mlp3<8>(stag, 128, 256, w0, w1, w2, b0, b1, b2, 128, lane);

  float* fst = (float*)stag;
  ln_stats(fst, muA, rsA, lane);

  for (int idx = lane; idx < 2048; idx += 32) {
    int row = idx >> 7, col = idx & 127;
    int64_t nid = n0 + row;
    if (nid < N) {
      float nn = (fst[idx] - muA[row]) * rsA[row] * lng[col] + lnb[col];
      float nh = node_h[nid*H + col] + nn;
      node_h[nid*H + col]  = nh;
      node_bf[nid*H + col] = (__bf16)nh;
    }
  }
}

// ---------------- Decoder: 128 -> 128 -> 128 -> 3, no LN ----------------
__global__ __launch_bounds__(256) void dec_kernel(
    const __bf16* __restrict__ node_bf, float* __restrict__ out,
    const __bf16* __restrict__ wgt, const float* b0, const float* b1, const float* b2, int N)
{
  extern __shared__ u32 lds[];
  const int tid = threadIdx.x, lane = tid & 31, wid = tid >> 5;
  for (int i = tid; i < DEC_WGT_U32/4; i += blockDim.x)
    ((uint4*)lds)[i] = ((const uint4*)wgt)[i];
  __syncthreads();
  const u32* w0 = lds;
  const u32* w1 = lds + (128*128)/2;
  const u32* w2 = w1 + (128*128)/2;   // [16][128], rows >=3 zero
  u32* stag = lds + DEC_WGT_U32 + wid * DEC_STG_U32;

  int64_t n0 = ((int64_t)blockIdx.x * 8 + wid) * 16;
  if (n0 >= N) return;

  for (int row = 0; row < 16; ++row) {
    int64_t nid = n0 + row; if (nid > N-1) nid = N-1;
    uint2 d = ((const uint2*)(node_bf + (size_t)nid * H))[lane];
    ((uint2*)(stag + row*64))[lane] = d;
  }
  WAIT_DS();

  mlp3<1>(stag, 64, 128, w0, w1, w2, b0, b1, b2, 3, lane);

  float* fst = (float*)stag;  // [16][16]
  for (int idx = lane; idx < 48; idx += 32) {
    int row = idx / 3, c = idx % 3;
    int64_t nid = n0 + row;
    if (nid < N) out[nid*3 + c] = fst[row*16 + c];
  }
}

// ---------------- Utility kernels ----------------
__global__ void zero_f32(float* p, long n) {
  long i = (long)blockIdx.x * blockDim.x + threadIdx.x;
  if (i < n) p[i] = 0.f;
}

// Transpose+pad+convert: w f32 [K][N] -> wT bf16 [Npad][Kpad]
__global__ void prep_w(const float* __restrict__ src, __bf16* __restrict__ dst,
                       int K, int N, int Kpad, int Npad) {
  int idx = blockIdx.x * blockDim.x + threadIdx.x;
  int total = Kpad * Npad;
  if (idx >= total) return;
  int n = idx / Kpad, k = idx % Kpad;
  float v = (k < K && n < N) ? src[(size_t)k * N + n] : 0.f;
  dst[idx] = (__bf16)v;
}

extern "C" void kernel_launch(void* const* d_in, const int* in_sizes, int n_in,
                              void* d_out, int out_size, void* d_ws, size_t ws_size,
                              hipStream_t stream) {
  (void)in_sizes; (void)n_in; (void)out_size; (void)ws_size;

  const float* x     = (const float*)d_in[0];
  const float* eattr = (const float*)d_in[1];
  const int*   eidx  = (const int*)d_in[2];
  const int* senders   = eidx;
  const int* receivers = eidx + NEDGES;
  auto P = [&](int i) { return (const float*)d_in[i]; };

  // Workspace carve
  char* ws = (char*)d_ws;
  size_t off = 0;
  auto carve = [&](size_t bytes) -> char* {
    char* p = ws + off;
    off = (off + bytes + 255) & ~(size_t)255;
    return p;
  };
  float*   node_h  = (float*)  carve((size_t)NNODES * H * 4);
  __bf16*  node_bf = (__bf16*) carve((size_t)NNODES * H * 2);
  float*   agg     = (float*)  carve((size_t)NNODES * H * 4);
  __bf16*  edge_bf = (__bf16*) carve((size_t)NEDGES * H * 2);
  __bf16*  wEncN   = (__bf16*) carve((size_t)36864 * 2);
  __bf16*  wEncE   = (__bf16*) carve((size_t)36864 * 2);
  __bf16*  wDec    = (__bf16*) carve((size_t)34816 * 2);
  __bf16*  wBlkE[4]; __bf16* wBlkN[4];
  for (int i = 0; i < 4; ++i) {
    wBlkE[i] = (__bf16*)carve((size_t)81920 * 2);
    wBlkN[i] = (__bf16*)carve((size_t)65536 * 2);
  }

  // Allow >64KB dynamic LDS
  hipFuncSetAttribute(reinterpret_cast<const void*>(edge_block_kernel),
                      hipFuncAttributeMaxDynamicSharedMemorySize, EDGE_LDS_BYTES);
  hipFuncSetAttribute(reinterpret_cast<const void*>(node_block_kernel),
                      hipFuncAttributeMaxDynamicSharedMemorySize, NODE_LDS_BYTES);
  hipFuncSetAttribute(reinterpret_cast<const void*>(enc_node_kernel),
                      hipFuncAttributeMaxDynamicSharedMemorySize, ENC_LDS_BYTES);
  hipFuncSetAttribute(reinterpret_cast<const void*>(enc_edge_kernel),
                      hipFuncAttributeMaxDynamicSharedMemorySize, ENC_LDS_BYTES);
  hipFuncSetAttribute(reinterpret_cast<const void*>(dec_kernel),
                      hipFuncAttributeMaxDynamicSharedMemorySize, DEC_LDS_BYTES);

  auto prep = [&](const float* src, __bf16* dst, int K, int N, int Kp, int Np) {
    int tot = Kp * Np;
    prep_w<<<(tot + 255) / 256, 256, 0, stream>>>(src, dst, K, N, Kp, Np);
  };

  // Param order (insertion order of setup_inputs): per MLP w0,b0,w1,b1,w2,b2[,ln_g,ln_b]
  // 3..10 enc_node, 11..18 enc_edge, 19..24 dec, 25+16i blocks[i].edge, +8 .node
  prep(P(3),  wEncN,                16, 128,  32, 128);
  prep(P(5),  wEncN + 32*128,      128, 128, 128, 128);
  prep(P(7),  wEncN + 32*128+16384,128, 128, 128, 128);
  prep(P(11), wEncE,                 4, 128,  32, 128);
  prep(P(13), wEncE + 32*128,      128, 128, 128, 128);
  prep(P(15), wEncE + 32*128+16384,128, 128, 128, 128);
  prep(P(19), wDec,                128, 128, 128, 128);
  prep(P(21), wDec + 16384,        128, 128, 128, 128);
  prep(P(23), wDec + 32768,        128,   3, 128,  16);
  for (int i = 0; i < 4; ++i) {
    int be = 25 + 16*i, bn = be + 8;
    prep(P(be+0), wBlkE[i],               384, 128, 384, 128);
    prep(P(be+2), wBlkE[i] + 49152,       128, 128, 128, 128);
    prep(P(be+4), wBlkE[i] + 49152+16384, 128, 128, 128, 128);
    prep(P(bn+0), wBlkN[i],               256, 128, 256, 128);
    prep(P(bn+2), wBlkN[i] + 32768,       128, 128, 128, 128);
    prep(P(bn+4), wBlkN[i] + 32768+16384, 128, 128, 128, 128);
  }

  const int nodeBlocks = ((NNODES + 15)/16 + 7) / 8;   // 391
  const int edgeBlocks = ((NEDGES + 15)/16 + 7) / 8;   // 6250

  enc_node_kernel<<<nodeBlocks, 256, ENC_LDS_BYTES, stream>>>(
      x, node_h, node_bf, wEncN, P(4), P(6), P(8), P(9), P(10), NNODES);
  enc_edge_kernel<<<edgeBlocks, 256, ENC_LDS_BYTES, stream>>>(
      eattr, edge_bf, wEncE, P(12), P(14), P(16), P(17), P(18), NEDGES);

  for (int i = 0; i < 4; ++i) {
    int be = 25 + 16*i, bn = be + 8;
    long aggN = (long)NNODES * H;
    zero_f32<<<(int)((aggN + 255) / 256), 256, 0, stream>>>(agg, aggN);
    edge_block_kernel<<<edgeBlocks, 256, EDGE_LDS_BYTES, stream>>>(
        node_bf, edge_bf, agg, senders, receivers, wBlkE[i],
        P(be+1), P(be+3), P(be+5), P(be+6), P(be+7), NEDGES);
    node_block_kernel<<<nodeBlocks, 256, NODE_LDS_BYTES, stream>>>(
        node_h, node_bf, agg, wBlkN[i],
        P(bn+1), P(bn+3), P(bn+5), P(bn+6), P(bn+7), NNODES);
  }

  dec_kernel<<<nodeBlocks, 256, DEC_LDS_BYTES, stream>>>(
      node_bf, (float*)d_out, wDec, P(20), P(22), P(24), NNODES);
}
